// TransformerBlock_1133871366380
// MI455X (gfx1250) — matrix operand order repaired and checked by reading, not compiled, not run on hardware
//
#include <hip/hip_runtime.h>
#include <cstdint>
#include <cstddef>

// ---------------------------------------------------------------- constants
static constexpr int EMB   = 512;
static constexpr int HEADS = 8;
static constexpr int HDIM  = 64;
static constexpr int FFN   = 2048;
static constexpr int SEQ   = 2048;
static constexpr int BATCH = 2;
static constexpr int TOK   = BATCH * SEQ;   // 4096

typedef __attribute__((ext_vector_type(16))) __bf16 bf16x16;
typedef __attribute__((ext_vector_type(8)))  __bf16 bf16x8;
typedef __attribute__((ext_vector_type(8)))  float  v8f;
typedef unsigned int u32x4 __attribute__((ext_vector_type(4)));
typedef int          i32x8 __attribute__((ext_vector_type(8)));
typedef int          i32x4 __attribute__((ext_vector_type(4)));

// -------- CDNA5 data-mover feature detection (degrade gracefully if absent)
#if __has_builtin(__builtin_amdgcn_global_load_async_to_lds_b128) && \
    __has_builtin(__builtin_amdgcn_s_wait_asynccnt)
#define ASYNC_LDS 1
#else
#define ASYNC_LDS 0
#endif
#if __has_builtin(__builtin_amdgcn_tensor_load_to_lds) && \
    __has_builtin(__builtin_amdgcn_s_wait_tensorcnt)
#define USE_TDM 1
#else
#define USE_TDM 0
#endif

// build AS1/AS3 v4i32 pointers via inttoptr (flat-LDS low 32 bits == LDS offset, ISA 10.2)
__device__ __forceinline__ __attribute__((address_space(1))) i32x4* gvec4(const void* p) {
  return (__attribute__((address_space(1))) i32x4*)(unsigned long long)(size_t)p;
}
__device__ __forceinline__ __attribute__((address_space(3))) i32x4* lvec4(void* p) {
  return (__attribute__((address_space(3))) i32x4*)(unsigned int)(size_t)p;
}

// move 16B global -> LDS; async DMA path when available (ASYNCcnt), else VGPR bounce
__device__ __forceinline__ void copy16(const __bf16* g, __bf16* l) {
#if ASYNC_LDS
  __builtin_amdgcn_global_load_async_to_lds_b128(gvec4(g), lvec4(l), 0, 0);
#else
  *(bf16x8*)l = *(const bf16x8*)g;
#endif
}
template <int N> __device__ __forceinline__ void wait_async() {
#if ASYNC_LDS
  __builtin_amdgcn_s_wait_asynccnt(N);
#endif
}
template <int N> __device__ __forceinline__ void wait_tensor() {
#if USE_TDM
  __builtin_amdgcn_s_wait_tensorcnt(N);
#endif
}

#if USE_TDM
// TDM: DMA a 64(K)x64(N) bf16 weight tile (row stride N halfs) into LDS with a
// 72-half padded row stride. Moved as 8B elements: tile 16x64, pad 4 DWORDs
// every 32 DWORDs (pad_interval code 4, pad_amount code 3). D# per ISA 8.3/8.4.
__device__ __forceinline__ void tdm_load_b_tile(const __bf16* gsrc, __bf16* ldst, int N) {
  const unsigned long long ga = (unsigned long long)(size_t)gsrc;
  const unsigned int       la = (unsigned int)(size_t)ldst;
  u32x4 g0;
  g0[0] = 1u;                                             // count=1, user mode
  g0[1] = la;                                             // lds_addr [63:32]
  g0[2] = (unsigned int)(ga & 0xFFFFFFFFull);             // global_addr lo
  g0[3] = (unsigned int)((ga >> 32) & 0x01FFFFFFull)      // global_addr hi (57b)
          | (2u << 30);                                   // type=2 ("image")
  i32x8 g1;
  g1[0] = (int)((3u << 16)      // data_size = 8B
              | (1u << 20)      // pad_enable
              | (4u << 22)      // pad_interval: 32 DWORDs
              | (3u << 25));    // pad_amount : 4 DWORDs
  g1[1] = (int)(16u << 16);     // tensor_dim0 = 16 elements (row = 128B)
  g1[2] = (int)(64u << 16);     // tensor_dim1 = 64 rows
  g1[3] = (int)(16u << 16);     // tile_dim0 = 16
  g1[4] = 64;                   // tile_dim1 = 64, tile_dim2 = 0
  g1[5] = N >> 2;               // tensor_dim0_stride = N*2B/8B elements
  g1[6] = 0;
  g1[7] = 0;
  i32x4 gz4; gz4[0] = 0; gz4[1] = 0; gz4[2] = 0; gz4[3] = 0;
  i32x8 gz8;
#pragma unroll
  for (int i = 0; i < 8; ++i) gz8[i] = 0;
  // 6-operand form (this toolchain): groups 0..3 + extra group + cpol
  __builtin_amdgcn_tensor_load_to_lds(g0, g1, gz4, gz4, gz8, 0);
}
#endif

// fp32 -> bf16 round-to-nearest-even (storage only)
__device__ __forceinline__ __bf16 f2bf(float f) {
  unsigned int u = __builtin_bit_cast(unsigned int, f);
  unsigned int r = u + 0x7FFFu + ((u >> 16) & 1u);
  unsigned short h = (unsigned short)(r >> 16);
  return __builtin_bit_cast(__bf16, h);
}

// ------------------------------------------------- WMMA fragment loaders (ISA 7.12.2)
__device__ __forceinline__ void load_a_frag(const __bf16* base, int ld,
                                            int row0, int k0, bf16x16& a, int lane) {
  const int g = lane >> 4, r = lane & 15;
  const __bf16* p = base + (size_t)(row0 + r) * ld + k0 + g * 16;
#pragma unroll
  for (int i = 0; i < 8; ++i) a[i] = p[i];
#pragma unroll
  for (int i = 0; i < 8; ++i) a[8 + i] = p[8 + i];
}
__device__ __forceinline__ void load_b_frag(const __bf16* base, int ld,
                                            int k0, int col0, bf16x16& b, int lane) {
  const int g = lane >> 4, c = lane & 15;
  const __bf16* p = base + (size_t)(k0 + g * 16) * ld + col0 + c;
#pragma unroll
  for (int i = 0; i < 16; ++i) b[i] = p[(size_t)i * ld];
}
__device__ __forceinline__ void zero8(v8f& v) {
#pragma unroll
  for (int e = 0; e < 8; ++e) v[e] = 0.0f;
}

// ---------------------------------------------------------------- cast f32 -> bf16
__global__ void cast_bf16_kernel(const float* __restrict__ in, __bf16* __restrict__ out, int n) {
  int i = blockIdx.x * blockDim.x + threadIdx.x;
  if (i < n) out[i] = f2bf(in[i]);
}

// ---------------------------------------------------------------- GEMM
// C[M,N] = A[M,K]*W[K,N] + bias. Macro tile 128x64, BK=64, double-buffered LDS.
// A via per-lane async-to-LDS, W via TDM (wave 0), overlapped with WMMA compute.
template <bool RELU, bool F32OUT>
__global__ __launch_bounds__(256)
void gemm_wmma(const __bf16* __restrict__ A, const __bf16* __restrict__ W,
               const float* __restrict__ bias, void* __restrict__ outp,
               int N, int K) {
  constexpr int LD = 72;                         // padded row stride (halfs)
  constexpr int ASZ = 128 * LD;                  // one A buffer
  constexpr int BSZ = 64 * LD;                   // one B buffer
  __shared__ __align__(16) __bf16 smem[2 * ASZ + 2 * BSZ];   // 55296 B

  const int tid  = threadIdx.x;
  const int lane = tid & 31;
  const int wave = tid >> 5;
  const int wm = wave >> 1, wn = wave & 1;       // 4x2 wave grid, 32x32 each
  const int m0 = blockIdx.y * 128;
  const int n0 = blockIdx.x * 64;

  const int ar = tid >> 1, acs = (tid & 1) * 32; // A loader: 32 halfs/thread (4x16B)
  const int br = tid >> 2, bcs = (tid & 3) * 16; // B fallback: 16 halfs/thread (2x16B)

#if USE_TDM
  constexpr int TOPS = 4;                        // async ops per tile per thread
#else
  constexpr int TOPS = 6;
#endif

  auto issue_tile = [&](int k0, int buf) {
    __bf16* Ab = smem + buf * ASZ;
    __bf16* Bb = smem + 2 * ASZ + buf * BSZ;
    const __bf16* asrc = A + (size_t)(m0 + ar) * K + k0 + acs;
#pragma unroll
    for (int u = 0; u < 4; ++u) copy16(asrc + u * 8, &Ab[ar * LD + acs + u * 8]);
#if USE_TDM
    if (wave == 0) tdm_load_b_tile(W + (size_t)k0 * N + n0, Bb, N);
#else
    const __bf16* bsrc = W + (size_t)(k0 + br) * N + n0 + bcs;
    copy16(bsrc,     &Bb[br * LD + bcs]);
    copy16(bsrc + 8, &Bb[br * LD + bcs + 8]);
#endif
  };

  v8f acc[2][2];
#pragma unroll
  for (int i = 0; i < 2; ++i)
#pragma unroll
    for (int j = 0; j < 2; ++j) zero8(acc[i][j]);

  const int nt = K >> 6;
  issue_tile(0, 0);
  for (int it = 0; it < nt; ++it) {
    if (it + 1 < nt) {                            // software pipeline: prefetch next
      issue_tile((it + 1) << 6, (it + 1) & 1);
      wait_async<TOPS>();
#if USE_TDM
      if (wave == 0) wait_tensor<1>();
#endif
    } else {
      wait_async<0>();
#if USE_TDM
      if (wave == 0) wait_tensor<0>();
#endif
    }
    __syncthreads();

    const __bf16* Ab = smem + (it & 1) * ASZ;
    const __bf16* Bb = smem + 2 * ASZ + (it & 1) * BSZ;
#pragma unroll
    for (int kk = 0; kk < 64; kk += 32) {
      bf16x16 a[2], b[2];
#pragma unroll
      for (int i = 0; i < 2; ++i) load_a_frag(Ab, LD, wm * 32 + i * 16, kk, a[i], lane);
#pragma unroll
      for (int j = 0; j < 2; ++j) load_b_frag(Bb, LD, kk, wn * 32 + j * 16, b[j], lane);
#pragma unroll
      for (int i = 0; i < 2; ++i)
#pragma unroll
        for (int j = 0; j < 2; ++j)
          acc[i][j] = __builtin_amdgcn_wmma_f32_16x16x32_bf16(
              false, a[i], false, b[j], (short)0, acc[i][j], false, false);
    }
    __syncthreads();
  }

  const int g = lane >> 4, c = lane & 15;
#pragma unroll
  for (int i = 0; i < 2; ++i)
#pragma unroll
    for (int j = 0; j < 2; ++j) {
      const int nn = n0 + wn * 32 + j * 16 + c;
      const float bv = bias[nn];
#pragma unroll
      for (int r = 0; r < 8; ++r) {
        const int mm = m0 + wm * 32 + i * 16 + g * 8 + r;
        float v = acc[i][j][r] + bv;
        if (RELU) v = fmaxf(v, 0.0f);
        if (F32OUT) ((float*)outp)[(size_t)mm * N + nn] = v;
        else        ((__bf16*)outp)[(size_t)mm * N + nn] = f2bf(v);
      }
    }
}

// ---------------------------------------------------------------- flash attention
// grid (SEQ/128, HEADS, BATCH), 256 threads; wave owns 16 query rows.
// Q fragments in registers; K/V 64-key blocks double-buffered via async-to-LDS.
__global__ __launch_bounds__(256)
void attn_kernel(const __bf16* __restrict__ Q, const __bf16* __restrict__ Km,
                 const __bf16* __restrict__ V, __bf16* __restrict__ ctx) {
  constexpr int LD  = 72;
  constexpr int KVS = 64 * LD;                   // one K or V buffer (4608 halfs)
  __shared__ __align__(16) __bf16 smem[4 * KVS + 8 * 16 * LD];  // 55296 B
  __bf16* Ps = smem + 4 * KVS;

  const int tid = threadIdx.x, lane = tid & 31, wave = tid >> 5;
  const int g = lane >> 4, c = lane & 15;
  const int qb = blockIdx.x, h = blockIdx.y, bb = blockIdx.z;
  const size_t tok0 = (size_t)bb * SEQ;
  const int hcol = h * HDIM;

  // stage Q tile 128x64 into LDS (region later reused by K/V buffers)
  {
    const int r = tid >> 1, ds = (tid & 1) * 32;
    const __bf16* src = Q + (tok0 + qb * 128 + r) * EMB + hcol + ds;
#pragma unroll
    for (int u = 0; u < 4; ++u)
      *(bf16x8*)&smem[r * LD + ds + u * 8] = *(const bf16x8*)(src + u * 8);
  }
  __syncthreads();
  bf16x16 aq[2];                                 // Q A-fragments live in VGPRs
#pragma unroll
  for (int d0 = 0; d0 < 2; ++d0)
    load_a_frag(smem + wave * 16 * LD, LD, 0, d0 * 32, aq[d0], lane);
  __syncthreads();                               // all Q reads done before overwrite

  const int kvr = tid >> 2, kvd = (tid & 3) * 16;
  auto issue_kv = [&](int kb, int buf) {
    __bf16* Kb = smem + (buf * 2 + 0) * KVS;
    __bf16* Vb = smem + (buf * 2 + 1) * KVS;
    const __bf16* ks = Km + (tok0 + kb * 64 + kvr) * EMB + hcol + kvd;
    const __bf16* vs = V  + (tok0 + kb * 64 + kvr) * EMB + hcol + kvd;
    copy16(ks,     &Kb[kvr * LD + kvd]);
    copy16(ks + 8, &Kb[kvr * LD + kvd + 8]);
    copy16(vs,     &Vb[kvr * LD + kvd]);
    copy16(vs + 8, &Vb[kvr * LD + kvd + 8]);
  };

  v8f accO[4];
#pragma unroll
  for (int j = 0; j < 4; ++j) zero8(accO[j]);
  float mrow[8], lrow[8];
#pragma unroll
  for (int r = 0; r < 8; ++r) { mrow[r] = -1e30f; lrow[r] = 0.0f; }

  const int kbmax = 2 * qb + 1;                  // causal block bound
  issue_kv(0, 0);
  for (int kb = 0; kb <= kbmax; ++kb) {
    if (kb < kbmax) { issue_kv(kb + 1, (kb + 1) & 1); wait_async<4>(); }
    else            { wait_async<0>(); }
    __syncthreads();
    const __bf16* Kb = smem + ((kb & 1) * 2 + 0) * KVS;
    const __bf16* Vb = smem + ((kb & 1) * 2 + 1) * KVS;

    // scores S = Q(16x64) x K^T(64x64): 4 N-tiles, 2 K-steps over head dim
    v8f s[4];
#pragma unroll
    for (int j = 0; j < 4; ++j) zero8(s[j]);
#pragma unroll
    for (int d0 = 0; d0 < 2; ++d0)
#pragma unroll
      for (int j = 0; j < 4; ++j) {
        bf16x16 bk;  // B[k=d][n=key] = K[key][d] -> contiguous per-lane reads
        const __bf16* p = Kb + (size_t)(j * 16 + c) * LD + d0 * 32 + g * 16;
#pragma unroll
        for (int i = 0; i < 16; ++i) bk[i] = p[i];
        s[j] = __builtin_amdgcn_wmma_f32_16x16x32_bf16(
            false, aq[d0], false, bk, (short)0, s[j], false, false);
      }

    // online softmax (row = g*8+r; reductions inside 16-lane half-groups)
    const int qrow_base = qb * 128 + wave * 16 + g * 8;
#pragma unroll
    for (int r = 0; r < 8; ++r) {
      const int q = qrow_base + r;
      float sv[4], smax = -1e30f;
#pragma unroll
      for (int j = 0; j < 4; ++j) {
        const int k = kb * 64 + j * 16 + c;
        float x = s[j][r] * 0.125f;                       // 1/sqrt(64)
        x += (k > q) ? -1e9f : (-0.05f * (float)(q - k)); // causal + recency
        sv[j] = x; smax = fmaxf(smax, x);
      }
#pragma unroll
      for (int mk = 1; mk < 16; mk <<= 1) smax = fmaxf(smax, __shfl_xor(smax, mk, 32));
      const float mnew = fmaxf(mrow[r], smax);
      const float corr = __expf(mrow[r] - mnew);
      float rsum = 0.0f;
#pragma unroll
      for (int j = 0; j < 4; ++j) {
        const float p = __expf(sv[j] - mnew);
        rsum += p;
        Ps[wave * 16 * LD + (g * 8 + r) * LD + j * 16 + c] = f2bf(p);
      }
#pragma unroll
      for (int mk = 1; mk < 16; mk <<= 1) rsum += __shfl_xor(rsum, mk, 32);
      lrow[r] = lrow[r] * corr + rsum;
      mrow[r] = mnew;
#pragma unroll
      for (int j2 = 0; j2 < 4; ++j2) accO[j2][r] = accO[j2][r] * corr;
    }

    // O += P(16x64) x V(64x64)
#pragma unroll
    for (int ks0 = 0; ks0 < 64; ks0 += 32) {
      bf16x16 ap;
      load_a_frag(Ps + wave * 16 * LD, LD, 0, ks0, ap, lane);
#pragma unroll
      for (int j2 = 0; j2 < 4; ++j2) {
        bf16x16 bv;
        load_b_frag(Vb, LD, ks0, j2 * 16, bv, lane);
        accO[j2] = __builtin_amdgcn_wmma_f32_16x16x32_bf16(
            false, ap, false, bv, (short)0, accO[j2], false, false);
      }
    }
    __syncthreads();
  }

#pragma unroll
  for (int j2 = 0; j2 < 4; ++j2)
#pragma unroll
    for (int r = 0; r < 8; ++r) {
      const float v = accO[j2][r] / lrow[r];
      const size_t t = tok0 + qb * 128 + wave * 16 + g * 8 + r;
      ctx[t * EMB + hcol + j2 * 16 + c] = f2bf(v);
    }
}

// ---------------------------------------------------------------- residual + LayerNorm
__global__ __launch_bounds__(256)
void ln_residual(const float* __restrict__ base, const float* __restrict__ delta,
                 const float* __restrict__ gamma, const float* __restrict__ beta,
                 float* __restrict__ outf, __bf16* __restrict__ outbf) {
  const int row  = blockIdx.x * 8 + (threadIdx.x >> 5);
  const int lane = threadIdx.x & 31;
  const float* pb = base  + (size_t)row * EMB;
  const float* pd = delta + (size_t)row * EMB;
  float vals[16], s = 0.0f, ss = 0.0f;
#pragma unroll
  for (int i = 0; i < 16; ++i) {
    const float v = pb[lane + i * 32] + pd[lane + i * 32];
    vals[i] = v; s += v; ss += v * v;
  }
#pragma unroll
  for (int mk = 1; mk < 32; mk <<= 1) { s += __shfl_xor(s, mk, 32); ss += __shfl_xor(ss, mk, 32); }
  const float mu   = s * (1.0f / EMB);
  const float var  = ss * (1.0f / EMB) - mu * mu;
  const float rstd = rsqrtf(var + 1e-5f);
#pragma unroll
  for (int i = 0; i < 16; ++i) {
    const int col = lane + i * 32;
    const float y = (vals[i] - mu) * rstd * gamma[col] + beta[col];
    outf[(size_t)row * EMB + col] = y;
    if (outbf != nullptr) outbf[(size_t)row * EMB + col] = f2bf(y);
  }
}

// ---------------------------------------------------------------- launcher
extern "C" void kernel_launch(void* const* d_in, const int* in_sizes, int n_in,
                              void* d_out, int out_size, void* d_ws, size_t ws_size,
                              hipStream_t stream) {
  (void)in_sizes; (void)n_in; (void)out_size; (void)ws_size;
  const float* x    = (const float*)d_in[0];
  const float* wq   = (const float*)d_in[1];
  const float* bq   = (const float*)d_in[2];
  const float* wk   = (const float*)d_in[3];
  const float* bk   = (const float*)d_in[4];
  const float* wv   = (const float*)d_in[5];
  const float* bv   = (const float*)d_in[6];
  const float* wo   = (const float*)d_in[7];
  const float* bo   = (const float*)d_in[8];
  const float* ln1g = (const float*)d_in[9];
  const float* ln1b = (const float*)d_in[10];
  const float* w1   = (const float*)d_in[11];
  const float* b1   = (const float*)d_in[12];
  const float* w2   = (const float*)d_in[13];
  const float* b2   = (const float*)d_in[14];
  const float* ln2g = (const float*)d_in[15];
  const float* ln2b = (const float*)d_in[16];
  float* out = (float*)d_out;

  char* wsp = (char*)d_ws;
  auto carve = [&](size_t bytes) -> char* {
    char* p = wsp; wsp += (bytes + 255) & ~(size_t)255; return p;
  };
  __bf16* xb   = (__bf16*)carve((size_t)TOK * EMB * 2);
  __bf16* wqb  = (__bf16*)carve((size_t)EMB * EMB * 2);
  __bf16* wkb  = (__bf16*)carve((size_t)EMB * EMB * 2);
  __bf16* wvb  = (__bf16*)carve((size_t)EMB * EMB * 2);
  __bf16* wob  = (__bf16*)carve((size_t)EMB * EMB * 2);
  __bf16* w1b  = (__bf16*)carve((size_t)EMB * FFN * 2);
  __bf16* w2b  = (__bf16*)carve((size_t)FFN * EMB * 2);
  __bf16* qw   = (__bf16*)carve((size_t)TOK * EMB * 2);
  __bf16* kw   = (__bf16*)carve((size_t)TOK * EMB * 2);
  __bf16* vw   = (__bf16*)carve((size_t)TOK * EMB * 2);
  __bf16* ctxb = (__bf16*)carve((size_t)TOK * EMB * 2);
  float*  tmpf = (float*) carve((size_t)TOK * EMB * 4);   // attn_out, then ffn_out
  float*  x1f  = (float*) carve((size_t)TOK * EMB * 4);
  __bf16* x1b  = (__bf16*)carve((size_t)TOK * EMB * 2);
  __bf16* hb   = (__bf16*)carve((size_t)TOK * FFN * 2);

  auto cast = [&](const float* src, __bf16* dst, int n) {
    cast_bf16_kernel<<<(n + 255) / 256, 256, 0, stream>>>(src, dst, n);
  };
  cast(x,  xb,  TOK * EMB);
  cast(wq, wqb, EMB * EMB);
  cast(wk, wkb, EMB * EMB);
  cast(wv, wvb, EMB * EMB);
  cast(wo, wob, EMB * EMB);
  cast(w1, w1b, EMB * FFN);
  cast(w2, w2b, FFN * EMB);

  const dim3 blk(256);
  const dim3 gE(EMB / 64, TOK / 128);
  const dim3 gF(FFN / 64, TOK / 128);

  gemm_wmma<false, false><<<gE, blk, 0, stream>>>(xb, wqb, bq, qw, EMB, EMB);
  gemm_wmma<false, false><<<gE, blk, 0, stream>>>(xb, wkb, bk, kw, EMB, EMB);
  gemm_wmma<false, false><<<gE, blk, 0, stream>>>(xb, wvb, bv, vw, EMB, EMB);

  attn_kernel<<<dim3(SEQ / 128, HEADS, BATCH), blk, 0, stream>>>(qw, kw, vw, ctxb);

  gemm_wmma<false, true><<<gE, blk, 0, stream>>>(ctxb, wob, bo, tmpf, EMB, EMB);
  ln_residual<<<TOK / 8, blk, 0, stream>>>(x, tmpf, ln1g, ln1b, x1f, x1b);

  gemm_wmma<true,  false><<<gF, blk, 0, stream>>>(x1b, w1b, b1, hb, FFN, EMB);
  gemm_wmma<false, true><<<gE, blk, 0, stream>>>(hb, w2b, b2, tmpf, EMB, FFN);
  ln_residual<<<TOK / 8, blk, 0, stream>>>(x1f, tmpf, ln2g, ln2b, out, (__bf16*)nullptr);
}
